// EncoderLayer_29111288332991
// MI455X (gfx1250) — compile-verified, matrix operand order checked by reading, not run on hardware
//
#include <hip/hip_runtime.h>
#include <hip/hip_bf16.h>
#include <stdint.h>

// ---------------------------------------------------------------------------
// Transformer encoder layer for MI455X (gfx1250, wave32, WMMA + TDM).
// GEMMs: tensor_load_to_lds (TDM) double-buffered LDS staging feeding
//        v_wmma_f32_16x16x32_f16 (f16 in, f32 accumulate).
// ---------------------------------------------------------------------------

#define BATCH   4
#define SEQ     1024
#define DMODEL  1024
#define HEADS   16
#define DKH     64
#define MROWS   (BATCH * SEQ)          // 4096

typedef __attribute__((ext_vector_type(16))) _Float16 v16h;
typedef __attribute__((ext_vector_type(8)))  _Float16 v8h;
typedef __attribute__((ext_vector_type(8)))  float    v8f;
typedef __attribute__((ext_vector_type(4)))  uint32_t u32x4;
typedef __attribute__((ext_vector_type(8)))  uint32_t u32x8;

union FragU { v16h v; v8h h[2]; _Float16 e[16]; };

// Load a 16x32 f16 fragment (A-style / B-as-transposed-style) from a
// row-major matrix. lane<16 holds row (row0+lane), K offsets {0..7,16..23};
// lane>=16 holds row (row0+lane-16), K offsets {8..15,24..31}.
static __device__ inline v16h load_frag(const _Float16* __restrict__ base,
                                        int row0, int ld, int k0) {
    const int lane = threadIdx.x & 31;
    const int r  = row0 + (lane & 15);
    const int kk = k0 + ((lane >> 4) << 3);
    const _Float16* p = base + (size_t)r * ld + kk;
    FragU f;
    f.h[0] = *(const v8h*)(p);
    f.h[1] = *(const v8h*)(p + 16);
    return f.v;
}

static __device__ inline v8f wmma_f16(v16h a, v16h b, v8f c) {
    // D = A(16x32 f16) * B(32x16 f16) + C(16x16 f32)
    return __builtin_amdgcn_wmma_f32_16x16x32_f16(
        /*neg_a=*/false, a, /*neg_b=*/false, b,
        /*c_mod=*/(short)0, c, /*reuse_a=*/false, /*reuse_b=*/false);
}

// ---------------------------------------------------------------------------
// Tensor Data Mover: DMA a 2D f16 tile (tile_rows x 32) from a row-major
// global matrix into LDS. Descriptor per CDNA5 ISA.
// Group2/3 NULL (2D tensor). Tracked with TENSORcnt.
// ---------------------------------------------------------------------------
static __device__ inline void tdm_load_tile(uint32_t lds_byte_addr,
                                            const _Float16* tile_ptr,
                                            uint32_t tile_rows,
                                            uint32_t row_stride_elems,
                                            uint32_t tensor_rows) {
    const uint64_t ga = (uint64_t)(uintptr_t)tile_ptr;
    u32x4 g0;
    g0[0] = 0x1u;                                   // count=1, user descriptor
    g0[1] = lds_byte_addr;                          // lds_addr
    g0[2] = (uint32_t)ga;                           // global_addr[31:0]
    g0[3] = (uint32_t)((ga >> 32) & 0x1FFFFFFu)     // global_addr[56:32]
          | (2u << 30);                             // type=2 (image)

    const uint32_t dim0  = row_stride_elems;        // tensor dim0 (elements)
    const uint32_t dim1  = tensor_rows;             // tensor dim1 (rows)
    const uint32_t tile0 = 32u;                     // tile width (elements)
    const uint32_t tile1 = tile_rows;               // tile height (rows)
    u32x8 g1;
    g1[0] = (1u << 16);                             // data_size=1 -> 2 bytes
    g1[1] = (dim0 & 0xFFFFu) << 16;                 // tensor_dim0[15:0]
    g1[2] = (dim0 >> 16) | ((dim1 & 0xFFFFu) << 16);// dim0[31:16], dim1[15:0]
    g1[3] = (dim1 >> 16) | (tile0 << 16);           // dim1[31:16], tile_dim0
    g1[4] = tile1;                                  // tile_dim1 (tile_dim2=0)
    g1[5] = row_stride_elems;                       // tensor_dim0_stride[31:0]
    g1[6] = 0u;                                     // stride[47:32], d1s[15:0]
    g1[7] = 0u;                                     // d1s[47:16]

    asm volatile("tensor_load_to_lds %0, %1"
                 :: "s"(g0), "s"(g1)
                 : "memory");
}

// ---------------------------------------------------------------------------
// Weight convert + transpose: W[k][n] fp32 -> Wt[n][k] f16
// ---------------------------------------------------------------------------
__global__ void wt_convert_kernel(const float* __restrict__ W,
                                  _Float16* __restrict__ Wt) {
    int idx = blockIdx.x * blockDim.x + threadIdx.x;      // coalesced read
    int k = idx >> 10;           // / DMODEL
    int n = idx & 1023;          // % DMODEL
    Wt[(size_t)n * DMODEL + k] = (_Float16)W[idx];
}

// ---------------------------------------------------------------------------
// LayerNorm over D=1024, one block (256 threads) per row.
// MODE 0: outH = f16(LN(x)), rawH = f16(x)
// MODE 1: outF = LN(x) fp32, outH = f16(LN(x))
// ---------------------------------------------------------------------------
template <int MODE>
__global__ void ln_kernel(const float* __restrict__ x,
                          const float* __restrict__ g,
                          const float* __restrict__ b,
                          _Float16* __restrict__ outH,
                          float* __restrict__ outF,
                          _Float16* __restrict__ rawH) {
    const int row = blockIdx.x;
    const int tid = threadIdx.x;
    const float* xr = x + (size_t)row * DMODEL;
    __shared__ float red[256];

    float vals[4];
    float s = 0.f;
#pragma unroll
    for (int i = 0; i < 4; ++i) { vals[i] = xr[tid + i * 256]; s += vals[i]; }
    red[tid] = s; __syncthreads();
    for (int st = 128; st > 0; st >>= 1) {
        if (tid < st) red[tid] += red[tid + st];
        __syncthreads();
    }
    const float mu = red[0] * (1.0f / DMODEL);
    __syncthreads();

    float vs = 0.f;
#pragma unroll
    for (int i = 0; i < 4; ++i) { float d = vals[i] - mu; vs += d * d; }
    red[tid] = vs; __syncthreads();
    for (int st = 128; st > 0; st >>= 1) {
        if (tid < st) red[tid] += red[tid + st];
        __syncthreads();
    }
    const float rstd = rsqrtf(red[0] * (1.0f / DMODEL) + 1e-5f);

#pragma unroll
    for (int i = 0; i < 4; ++i) {
        int c = tid + i * 256;
        float y = (vals[i] - mu) * rstd * g[c] + b[c];
        size_t o = (size_t)row * DMODEL + c;
        if (MODE == 0) {
            outH[o] = (_Float16)y;
            rawH[o] = (_Float16)vals[i];
        } else {
            outF[o] = y;
            outH[o] = (_Float16)y;
        }
    }
}

// ---------------------------------------------------------------------------
// GEMM: C[M,N] = A[M,K]@W[K,N] + bias, A f16 row-major, Wt f16 [N,K].
// Block = 256 threads (8 waves). Block tile 128x64; wave tile 32x32.
// TDM double-buffers A(128x32) and B(64x32) tiles in LDS; wave 0 drives DMA.
// MODE 0: store f16        MODE 1: outF = residual + C (fp32)
// MODE 2: store relu() f16
// ---------------------------------------------------------------------------
template <int MODE>
__global__ void gemm_wmma_kernel(const _Float16* __restrict__ A,
                                 const _Float16* __restrict__ Wt,
                                 const float* __restrict__ bias,
                                 const float* __restrict__ residual,
                                 float* __restrict__ outF,
                                 _Float16* __restrict__ outH,
                                 int M, int N, int K) {
    __shared__ _Float16 ldsA[2][128 * 32];   // 16 KiB
    __shared__ _Float16 ldsB[2][64 * 32];    //  8 KiB

    const int w    = threadIdx.x >> 5;
    const int wmof = (w & 3) * 32;
    const int wnof = (w >> 2) * 32;
    const int bm   = blockIdx.x * 128;
    const int bn   = blockIdx.y * 64;
    const int lane = threadIdx.x & 31;
    const int cn   = lane & 15;
    const int half = lane >> 4;

    const uint32_t ldsA0 = (uint32_t)(uintptr_t)(void*)&ldsA[0][0];
    const uint32_t ldsA1 = (uint32_t)(uintptr_t)(void*)&ldsA[1][0];
    const uint32_t ldsB0 = (uint32_t)(uintptr_t)(void*)&ldsB[0][0];
    const uint32_t ldsB1 = (uint32_t)(uintptr_t)(void*)&ldsB[1][0];

    if (w == 0) {   // prime the pipeline: buffer 0 <- k0 = 0
        tdm_load_tile(ldsA0, A  + (size_t)bm * K, 128, (uint32_t)K, (uint32_t)M);
        tdm_load_tile(ldsB0, Wt + (size_t)bn * K,  64, (uint32_t)K, (uint32_t)N);
    }

    v8f acc00 = {}, acc01 = {}, acc10 = {}, acc11 = {};

    for (int k0 = 0; k0 < K; k0 += 32) {
        const int cur = (k0 >> 5) & 1;
        if (w == 0) {
            if (k0 + 32 < K) {   // prefetch next pair into the other buffer
                const uint32_t nA = cur ? ldsA0 : ldsA1;
                const uint32_t nB = cur ? ldsB0 : ldsB1;
                tdm_load_tile(nA, A  + (size_t)bm * K + (k0 + 32), 128, (uint32_t)K, (uint32_t)M);
                tdm_load_tile(nB, Wt + (size_t)bn * K + (k0 + 32),  64, (uint32_t)K, (uint32_t)N);
                // in-order TENSORcnt: <=2 outstanding => current pair landed
                __builtin_amdgcn_s_wait_tensorcnt(2);
            } else {
                __builtin_amdgcn_s_wait_tensorcnt(0);
            }
        }
        __syncthreads();                         // publish current tiles

        v16h a0 = load_frag(&ldsA[cur][0], wmof,      32, 0);
        v16h a1 = load_frag(&ldsA[cur][0], wmof + 16, 32, 0);
        v16h b0 = load_frag(&ldsB[cur][0], wnof,      32, 0);
        v16h b1 = load_frag(&ldsB[cur][0], wnof + 16, 32, 0);
        acc00 = wmma_f16(a0, b0, acc00);
        acc01 = wmma_f16(a0, b1, acc01);
        acc10 = wmma_f16(a1, b0, acc10);
        acc11 = wmma_f16(a1, b1, acc11);

        __syncthreads();                         // reads done before overwrite
    }

    const v8f* accs[4] = { &acc00, &acc01, &acc10, &acc11 };
#pragma unroll
    for (int t = 0; t < 4; ++t) {
        const int mi = t >> 1, ni = t & 1;
#pragma unroll
        for (int i = 0; i < 8; ++i) {
            int r = bm + wmof + mi * 16 + i + 8 * half;
            int c = bn + wnof + ni * 16 + cn;
            float v = (*accs[t])[i] + bias[c];
            size_t o = (size_t)r * N + c;
            if (MODE == 0) {
                outH[o] = (_Float16)v;
            } else if (MODE == 1) {
                outF[o] = residual[o] + v;
            } else {
                v = v > 0.f ? v : 0.f;
                outH[o] = (_Float16)v;
            }
        }
    }
}

// ---------------------------------------------------------------------------
// Transpose V: v_h[(b*S+s), h*DKH+d] -> vT[((b*H+h)*DKH+d)*S + s]
// ---------------------------------------------------------------------------
__global__ void vtrans_kernel(const _Float16* __restrict__ v_h,
                              _Float16* __restrict__ vT) {
    int idx = blockIdx.x * blockDim.x + threadIdx.x;   // over MROWS*DMODEL
    int row = idx >> 10;            // b*S + s
    int col = idx & 1023;           // h*DKH + d
    int b = row >> 10, s = row & 1023;
    int h = col >> 6,  d = col & 63;
    vT[(size_t)((b * HEADS + h) * DKH + d) * SEQ + s] = v_h[idx];
}

// ---------------------------------------------------------------------------
// Flash attention. Grid: B*H*(S/128) blocks of 256 threads (8 waves).
// Each wave: one 16-row query tile, streams 32-key blocks.
// ---------------------------------------------------------------------------
__global__ void attention_kernel(const _Float16* __restrict__ qh,
                                 const _Float16* __restrict__ kh,
                                 const _Float16* __restrict__ vT,
                                 const float* __restrict__ src_mask,
                                 const unsigned char* __restrict__ kpm,
                                 _Float16* __restrict__ ctx) {
    const int bid  = blockIdx.x;
    const int qb   = bid & 7;            // S/128 = 8 q-blocks
    const int h    = (bid >> 3) & (HEADS - 1);
    const int b    = bid >> 7;
    const int wave = threadIdx.x >> 5;
    const int lane = threadIdx.x & 31;
    const int cn   = lane & 15;
    const int half = lane >> 4;
    const int q0   = qb * 128 + wave * 16;

    const _Float16* qbase = qh + (size_t)(b * SEQ) * DMODEL + h * DKH;
    const _Float16* kbase = kh + (size_t)(b * SEQ) * DMODEL + h * DKH;
    const _Float16* vbase = vT + (size_t)((b * HEADS + h) * DKH) * SEQ;

    __shared__ _Float16 pl[8][16 * 32];   // per-wave P staging (C->A relayout)

    const v16h qa0 = load_frag(qbase, q0, DMODEL, 0);
    const v16h qa1 = load_frag(qbase, q0, DMODEL, 32);

    float run_m[8], run_l[8];
    v8f o0 = {}, o1 = {}, o2 = {}, o3 = {};
#pragma unroll
    for (int i = 0; i < 8; ++i) { run_m[i] = -1e30f; run_l[i] = 0.f; }

    for (int kb = 0; kb < SEQ; kb += 32) {
        // scores: 16x32 tile = two 16x16 WMMA tiles, K=64 in two steps
        v8f s0 = {}, s1 = {};
        s0 = wmma_f16(qa0, load_frag(kbase, kb,      DMODEL, 0),  s0);
        s0 = wmma_f16(qa1, load_frag(kbase, kb,      DMODEL, 32), s0);
        s1 = wmma_f16(qa0, load_frag(kbase, kb + 16, DMODEL, 0),  s1);
        s1 = wmma_f16(qa1, load_frag(kbase, kb + 16, DMODEL, 32), s1);

        const int key0 = kb + cn, key1 = kb + 16 + cn;
        const float mk0 = kpm[b * SEQ + key0] ? -1e30f : 0.f;
        const float mk1 = kpm[b * SEQ + key1] ? -1e30f : 0.f;

        float fac[8];
#pragma unroll
        for (int i = 0; i < 8; ++i) {
            const int r = q0 + i + 8 * half;
            float sc0 = s0[i] * 0.125f + src_mask[(size_t)r * SEQ + key0] + mk0;
            float sc1 = s1[i] * 0.125f + src_mask[(size_t)r * SEQ + key1] + mk1;
            // row max over 16 lanes of this half-wave
            float tm = fmaxf(sc0, sc1);
            for (int m = 1; m < 16; m <<= 1) tm = fmaxf(tm, __shfl_xor(tm, m, 16));
            const float nm = fmaxf(run_m[i], tm);
            fac[i] = __expf(run_m[i] - nm);
            run_m[i] = nm;
            const float p0 = __expf(sc0 - nm);
            const float p1 = __expf(sc1 - nm);
            pl[wave][(i + 8 * half) * 32 + cn]      = (_Float16)p0;
            pl[wave][(i + 8 * half) * 32 + 16 + cn] = (_Float16)p1;
            float rs = p0 + p1;
            for (int m = 1; m < 16; m <<= 1) rs += __shfl_xor(rs, m, 16);
            run_l[i] = run_l[i] * fac[i] + rs;
        }
#pragma unroll
        for (int i = 0; i < 8; ++i) {
            o0[i] *= fac[i]; o1[i] *= fac[i]; o2[i] *= fac[i]; o3[i] *= fac[i];
        }
        __syncthreads();                        // publish P tile
        const v16h pf = load_frag(&pl[wave][0], 0, 32, 0);
        o0 = wmma_f16(pf, load_frag(vbase, 0,  SEQ, kb), o0);
        o1 = wmma_f16(pf, load_frag(vbase, 16, SEQ, kb), o1);
        o2 = wmma_f16(pf, load_frag(vbase, 32, SEQ, kb), o2);
        o3 = wmma_f16(pf, load_frag(vbase, 48, SEQ, kb), o3);
        __syncthreads();                        // protect LDS before next write
    }

#pragma unroll
    for (int i = 0; i < 8; ++i) {
        const int r = q0 + i + 8 * half;
        const float inv = 1.f / run_l[i];
        const size_t rb = (size_t)(b * SEQ + r) * DMODEL + h * DKH;
        ctx[rb + cn]      = (_Float16)(o0[i] * inv);
        ctx[rb + 16 + cn] = (_Float16)(o1[i] * inv);
        ctx[rb + 32 + cn] = (_Float16)(o2[i] * inv);
        ctx[rb + 48 + cn] = (_Float16)(o3[i] * inv);
    }
}

// ---------------------------------------------------------------------------
// Launch
// ---------------------------------------------------------------------------
extern "C" void kernel_launch(void* const* d_in, const int* in_sizes, int n_in,
                              void* d_out, int out_size, void* d_ws, size_t ws_size,
                              hipStream_t stream) {
    const float* x        = (const float*)d_in[0];
    const float* src_mask = (const float*)d_in[1];
    const unsigned char* kpm = (const unsigned char*)d_in[2];
    const float* g1  = (const float*)d_in[3];
    const float* b1n = (const float*)d_in[4];
    const float* Wq = (const float*)d_in[5];  const float* bq = (const float*)d_in[6];
    const float* Wk = (const float*)d_in[7];  const float* bk = (const float*)d_in[8];
    const float* Wv = (const float*)d_in[9];  const float* bv = (const float*)d_in[10];
    const float* Wo = (const float*)d_in[11]; const float* bo = (const float*)d_in[12];
    const float* W1 = (const float*)d_in[13]; const float* b1 = (const float*)d_in[14];
    const float* W2 = (const float*)d_in[15]; const float* b2 = (const float*)d_in[16];
    float* out = (float*)d_out;

    const size_t MD_h = (size_t)MROWS * DMODEL * sizeof(_Float16);  // 8 MiB
    const size_t MD_f = (size_t)MROWS * DMODEL * sizeof(float);     // 16 MiB
    const size_t DD_h = (size_t)DMODEL * DMODEL * sizeof(_Float16); // 2 MiB

    char* p = (char*)d_ws;
    _Float16* x_h   = (_Float16*)p; p += MD_h;
    _Float16* xq_h  = (_Float16*)p; p += MD_h;
    _Float16* wt[6];
    for (int i = 0; i < 6; ++i) { wt[i] = (_Float16*)p; p += DD_h; }
    _Float16* q_h   = (_Float16*)p; p += MD_h;
    _Float16* k_h   = (_Float16*)p; p += MD_h;
    _Float16* v_h   = (_Float16*)p; p += MD_h;
    _Float16* vT    = (_Float16*)p; p += MD_h;
    _Float16* ctx_h = (_Float16*)p; p += MD_h;
    float*    xres  = (float*)p;    p += MD_f;
    float*    x2f   = (float*)p;    p += MD_f;
    _Float16* x2h   = (_Float16*)p; p += MD_h;
    _Float16* ff1_h = (_Float16*)p; p += MD_h;

    const int cvt_blocks = (DMODEL * DMODEL) / 256;
    wt_convert_kernel<<<cvt_blocks, 256, 0, stream>>>(Wq, wt[0]);
    wt_convert_kernel<<<cvt_blocks, 256, 0, stream>>>(Wk, wt[1]);
    wt_convert_kernel<<<cvt_blocks, 256, 0, stream>>>(Wv, wt[2]);
    wt_convert_kernel<<<cvt_blocks, 256, 0, stream>>>(Wo, wt[3]);
    wt_convert_kernel<<<cvt_blocks, 256, 0, stream>>>(W1, wt[4]);
    wt_convert_kernel<<<cvt_blocks, 256, 0, stream>>>(W2, wt[5]);

    // LN1: xq_h = f16(LN(x)); x_h = f16(x)
    ln_kernel<0><<<MROWS, 256, 0, stream>>>(x, g1, b1n, xq_h, nullptr, x_h);

    dim3 ggrid(MROWS / 128, DMODEL / 64);
    // Q/K/V projections (f16 outputs)
    gemm_wmma_kernel<0><<<ggrid, 256, 0, stream>>>(xq_h, wt[0], bq, nullptr, nullptr, q_h, MROWS, DMODEL, DMODEL);
    gemm_wmma_kernel<0><<<ggrid, 256, 0, stream>>>(x_h,  wt[1], bk, nullptr, nullptr, k_h, MROWS, DMODEL, DMODEL);
    gemm_wmma_kernel<0><<<ggrid, 256, 0, stream>>>(x_h,  wt[2], bv, nullptr, nullptr, v_h, MROWS, DMODEL, DMODEL);

    vtrans_kernel<<<(MROWS * DMODEL) / 256, 256, 0, stream>>>(v_h, vT);

    // attention -> ctx_h
    attention_kernel<<<BATCH * HEADS * (SEQ / 128), 256, 0, stream>>>(
        q_h, k_h, vT, src_mask, kpm, ctx_h);

    // xres = x + ctx @ Wo + bo
    gemm_wmma_kernel<1><<<ggrid, 256, 0, stream>>>(ctx_h, wt[3], bo, x, xres, nullptr, MROWS, DMODEL, DMODEL);

    // LN2: x2f = LN(xres) fp32, x2h = f16 copy
    ln_kernel<1><<<MROWS, 256, 0, stream>>>(xres, g1, b1n, x2h, x2f, nullptr);

    // ff1 = relu(x2 @ W1 + b1)
    gemm_wmma_kernel<2><<<ggrid, 256, 0, stream>>>(x2h, wt[4], b1, nullptr, nullptr, ff1_h, MROWS, DMODEL, DMODEL);

    // out = x2 + ff1 @ W2 + b2
    gemm_wmma_kernel<1><<<ggrid, 256, 0, stream>>>(ff1_h, wt[5], b2, x2f, out, nullptr, MROWS, DMODEL, DMODEL);
}
